// TimeLSTM_53489522704624
// MI455X (gfx1250) — matrix operand order repaired and checked by reading, not compiled, not run on hardware
//
#include <hip/hip_runtime.h>

typedef __attribute__((ext_vector_type(16))) _Float16 v16h;
typedef __attribute__((ext_vector_type(8)))  _Float16 v8h;
typedef __attribute__((ext_vector_type(8)))  float    v8f;

#define B_ 32
#define T_ 128
#define F_ 64
#define H_ 64
#define AS 72   // f16 mirror row stride in halves: 144 B/row, 16 B aligned, conflict-free

__device__ __forceinline__ float sigmoidf_(float x) { return 1.0f / (1.0f + __expf(-x)); }
__device__ __forceinline__ float tanhf_(float x)    { return 1.0f - 2.0f / (__expf(2.0f * x) + 1.0f); }
__device__ __forceinline__ float eluf_(float x)     { return x > 0.0f ? x : (__expf(x) - 1.0f); }

// A-fragment (16x32 f16) for M-tile mt, K-chunk kc from the f16 LDS mirror.
// ISA 7.12.2 A layout: lane<16 -> M=lane, halves 0-7 = K[base..base+7], halves 8-15 = K[base+16..base+23],
// base = kc*32 + 8*hiHalf  ->  two contiguous 16-byte runs per lane -> 2x ds_load_b128.
__device__ __forceinline__ v16h mkAfragH(const _Float16* mir, int mt, int kc, int lane) {
    const int m   = 16 * mt + (lane & 15);
    const int hi  = (lane >> 4) & 1;
    const int off = m * AS + kc * 32 + 8 * hi;
    const v8h lo = *(const v8h*)(mir + off);
    const v8h hh = *(const v8h*)(mir + off + 16);
    v16h a;
#pragma unroll
    for (int i = 0; i < 8; ++i) { a[i] = lo[i]; a[8 + i] = hh[i]; }
    return a;
}

// B-fragment (32x16 f16) for N-tile w, K-chunk kc, from global f32 row-major W[k][n].
// B layout (matches A): halves 0-7 = K[8*hi..8*hi+7], halves 8-15 = K[16+8*hi..16+8*hi+7].
__device__ __forceinline__ v16h mkBfrag(const float* __restrict__ W, int w, int kc, int lane) {
    const int n  = 16 * w + (lane & 15);
    const int hi = (lane >> 4) & 1;
    v16h b;
#pragma unroll
    for (int i = 0; i < 16; ++i) {
        const int k = kc * 32 + (i & 7) + 8 * hi + 16 * (i >> 3);
        b[i] = (_Float16)W[k * H_ + n];
    }
    return b;
}

__device__ __forceinline__ v8f wmma_acc(v16h a, v16h b, v8f c) {
    return __builtin_amdgcn_wmma_f32_16x16x32_f16(false, a, false, b, (short)0, c, false, false);
}
__device__ __forceinline__ v8f gemm2(v16h a0, v16h a1, v16h b0, v16h b1) {
    v8f c = {0.f, 0.f, 0.f, 0.f, 0.f, 0.f, 0.f, 0.f};
    c = wmma_acc(a0, b0, c);
    c = wmma_acc(a1, b1, c);
    return c;
}

// Precompute fw[b,s] = sum_f exp(-0.5 * freqs[b,s,f])  (state-independent coupling term)
__global__ void fw_precompute_kernel(const float* __restrict__ freqs, float* __restrict__ fwbt) {
    const int idx = blockIdx.x * blockDim.x + threadIdx.x;   // idx = b*T + s
    if (idx >= B_ * T_) return;
    const float* p = freqs + (size_t)idx * F_;
    float s = 0.0f;
#pragma unroll
    for (int ff = 0; ff < F_; ++ff) s += __expf(-0.5f * p[ff]);
    fwbt[idx] = s;
}

__global__ __launch_bounds__(256, 1)
void timelstm_kernel(const float* __restrict__ inputs, const float* __restrict__ times,
                     const float* __restrict__ lastv,  const float* __restrict__ fwbt,
                     const float* __restrict__ Uj, const float* __restrict__ Ui,
                     const float* __restrict__ Uf, const float* __restrict__ Uo,
                     const float* __restrict__ Uc, const float* __restrict__ Ulast,
                     const float* __restrict__ Utime, const float* __restrict__ Dw,
                     const float* __restrict__ Wj, const float* __restrict__ Wi,
                     const float* __restrict__ Wf, const float* __restrict__ Wo,
                     const float* __restrict__ Wc, const float* __restrict__ Wd,
                     const float* __restrict__ Wdec,
                     const float* __restrict__ Wci, const float* __restrict__ Wcf,
                     const float* __restrict__ Wco,
                     const float* __restrict__ bj, const float* __restrict__ bi,
                     const float* __restrict__ bfv, const float* __restrict__ bo,
                     const float* __restrict__ bc, const float* __restrict__ blast,
                     const float* __restrict__ btime, const float* __restrict__ bd,
                     float* __restrict__ out)
{
    __shared__ __align__(16) _Float16 hM [B_ * AS];   // f16 mirror of h
    __shared__ __align__(16) _Float16 cM [B_ * AS];   // f16 mirror of c
    __shared__ __align__(16) _Float16 h2M[B_ * AS];   // f16 mirror of h2
    __shared__ float rowX[B_], rowTm[B_], rowSTm[B_], rowL[B_], rowFw[B_];

    const int tid  = threadIdx.x;
    const int lane = tid & 31;
    const int wave = tid >> 5;            // 8 waves
    const int w    = wave & 3;            // N-tile
    const int mt   = wave >> 2;           // M-tile
    const int f    = blockIdx.x;          // feature
    const int dir  = blockIdx.y;          // 0 = fwd, 1 = bwd
    const int hi   = (lane >> 4) & 1;
    const int n    = 16 * w + (lane & 15);    // owned output column
    const int m0   = 16 * mt + 8 * hi;        // first owned row (rows m0..m0+7)

    // per-lane column-n constants for feature f
    const int fo = f * H_ + n;
    const float uj = Uj[fo],  ui = Ui[fo],  uf = Uf[fo],  uo = Uo[fo];
    const float uc = Uc[fo],  ul = Ulast[fo], ut = Utime[fo], dw = Dw[fo];
    const float wci = Wci[fo], wcf = Wcf[fo], wco = Wco[fo];
    const float vbj = bj[fo], vbi = bi[fo], vbf = bfv[fo], vbo = bo[fo];
    const float vbc = bc[fo], vbl = blast[fo], vbt = btime[fo], vbd = bd[fo];

    // persistent f16 B-fragments for the 7 weight matrices (this wave's N-tile)
    const size_t wb = (size_t)f * H_ * H_;
    const v16h Bj0 = mkBfrag(Wj  + wb, w, 0, lane), Bj1 = mkBfrag(Wj  + wb, w, 1, lane);
    const v16h Bi0 = mkBfrag(Wi  + wb, w, 0, lane), Bi1 = mkBfrag(Wi  + wb, w, 1, lane);
    const v16h Bf0 = mkBfrag(Wf  + wb, w, 0, lane), Bf1 = mkBfrag(Wf  + wb, w, 1, lane);
    const v16h Bo0 = mkBfrag(Wo  + wb, w, 0, lane), Bo1 = mkBfrag(Wo  + wb, w, 1, lane);
    const v16h Bc0 = mkBfrag(Wc  + wb, w, 0, lane), Bc1 = mkBfrag(Wc  + wb, w, 1, lane);
    const v16h Bd0 = mkBfrag(Wd  + wb, w, 0, lane), Bd1 = mkBfrag(Wd  + wb, w, 1, lane);
    const v16h Bq0 = mkBfrag(Wdec+ wb, w, 0, lane), Bq1 = mkBfrag(Wdec+ wb, w, 1, lane);

    // persistent f32 recurrent state (this wave's (m,n) slice) + zero mirrors
    v8f hReg = {0.f,0.f,0.f,0.f,0.f,0.f,0.f,0.f};
    v8f cReg = {0.f,0.f,0.f,0.f,0.f,0.f,0.f,0.f};
    for (int i = tid; i < B_ * AS; i += 256) {
        hM[i] = (_Float16)0.f; cM[i] = (_Float16)0.f; h2M[i] = (_Float16)0.f;
    }
    __syncthreads();

    for (int s = 0; s < T_; ++s) {
        // ---- stage per-row scalars (rows = batch) ----
        if (tid < B_) {
            const int b  = tid;
            const int tx = dir ? (T_ - 1 - s) : s;                    // x index
            const int tt = dir ? ((s == 0) ? 0 : (T_ - s)) : s;       // t index (shifted flip)
            const float tv = times[(b * T_ + tt) * F_ + f];
            const float tm = 1.0f / __logf(tv + 2.7182818284590452f);
            rowX[b]   = inputs[(b * T_ + tx) * F_ + f];
            rowTm[b]  = tm;
            rowSTm[b] = sigmoidf_(tm);
            rowL[b]   = lastv[(b * T_ + s) * F_ + f];
            rowFw[b]  = fwbt[b * T_ + s];
        }
        __syncthreads();

        // ---- GEMMs on old state: D_ST_raw = c @ Wdecomp ; fdh_raw = h @ Wd ----
        v16h a0 = mkAfragH(cM, mt, 0, lane), a1 = mkAfragH(cM, mt, 1, lane);
        const v8f dst = gemm2(a0, a1, Bq0, Bq1);
        a0 = mkAfragH(hM, mt, 0, lane); a1 = mkAfragH(hM, mt, 1, lane);
        const v8f fdh = gemm2(a0, a1, Bd0, Bd1);

        // ---- phase-1 elementwise: c2, last_t, h2 (state read from registers) ----
        v8f c2v, ltv;
#pragma unroll
        for (int r = 0; r < 8; ++r) {
            const int m = m0 + r;
            const float tm   = rowTm[m];
            const float fwv  = rowFw[m];
            const float lv   = rowL[m];
            const float dstv = tanhf_(dst[r]);
            const float fde  = sigmoidf_(fwv * dw + fdh[r] + vbd);
            const float ddec = dstv * (tm * fde);
            c2v[r] = cReg[r] - dstv + 2.0f * ddec;
            const float lte = eluf_(lv * ul + vbl);
            ltv[r] = lte;
            h2M[m * AS + n] = (_Float16)(hReg[r] + lte);
        }
        __syncthreads();

        // ---- 6 GEMMs on h2 ----
        a0 = mkAfragH(h2M, mt, 0, lane); a1 = mkAfragH(h2M, mt, 1, lane);
        const v8f gj = gemm2(a0, a1, Bj0, Bj1);
        const v8f gd = gemm2(a0, a1, Bd0, Bd1);
        const v8f gi = gemm2(a0, a1, Bi0, Bi1);
        const v8f gf = gemm2(a0, a1, Bf0, Bf1);
        const v8f gc = gemm2(a0, a1, Bc0, Bc1);
        const v8f go = gemm2(a0, a1, Bo0, Bo1);

        // ---- phase-2 elementwise: gates, Ct, h_new; write output ----
        const int tout = dir ? (T_ - 1 - s) : s;
#pragma unroll
        for (int r = 0; r < 8; ++r) {
            const int m = m0 + r;
            const float xv  = rowX[m];
            const float tm  = rowTm[m];
            const float stm = rowSTm[m];
            const float fwv = rowFw[m];
            const float c2e = c2v[r];
            const float lte = ltv[r];

            const float jv   = tanhf_(gj[r] + xv * uj + vbj);
            const float tg   = sigmoidf_(xv * ut + stm + vbt);
            const float fd2  = sigmoidf_(fwv * dw + gd[r] + vbd);
            const float iv   = sigmoidf_(xv * ui + gi[r] + c2e * wci + vbi * fd2);
            const float fv   = sigmoidf_(xv * uf + gf[r] + c2e * wcf + vbf + jv);
            const float fnew = fv * tm + (1.0f - fv) * fd2;
            const float cg   = tanhf_(xv * uc + gc[r] + vbc);
            const float ct   = (fnew + tg) * c2e + iv * jv * tg * cg;
            const float ov   = sigmoidf_(xv * uo + go[r] + tg + lte + ct * wco + vbo);
            const float hn   = ov * tanhf_(ct + lte);

            hReg[r] = hn;
            cReg[r] = ct;
            out[(((size_t)tout * B_ + m) * F_ + f) * (2 * H_) + dir * H_ + n] = hn;
        }
        __syncthreads();   // all reads of hM/cM/h2M done before mirror overwrite

#pragma unroll
        for (int r = 0; r < 8; ++r) {
            const int m = m0 + r;
            hM[m * AS + n] = (_Float16)hReg[r];
            cM[m * AS + n] = (_Float16)cReg[r];
        }
        // next iteration's top barrier (after row-scalar staging) guards these writes
    }
}

extern "C" void kernel_launch(void* const* d_in, const int* in_sizes, int n_in,
                              void* d_out, int out_size, void* d_ws, size_t ws_size,
                              hipStream_t stream) {
    const float* inputs = (const float*)d_in[0];
    const float* times  = (const float*)d_in[1];
    const float* lastv  = (const float*)d_in[2];
    const float* freqs  = (const float*)d_in[3];
    const float* Uj   = (const float*)d_in[4];
    const float* Ui   = (const float*)d_in[5];
    const float* Uf   = (const float*)d_in[6];
    const float* Uo   = (const float*)d_in[7];
    const float* Uc   = (const float*)d_in[8];
    const float* Ulast= (const float*)d_in[9];
    const float* Utime= (const float*)d_in[10];
    const float* Dw   = (const float*)d_in[11];
    const float* Wj   = (const float*)d_in[12];
    const float* Wi   = (const float*)d_in[13];
    const float* Wf   = (const float*)d_in[14];
    const float* Wo   = (const float*)d_in[15];
    const float* Wc   = (const float*)d_in[16];
    const float* Wd   = (const float*)d_in[17];
    const float* Wdec = (const float*)d_in[18];
    const float* Wci  = (const float*)d_in[19];
    const float* Wcf  = (const float*)d_in[20];
    const float* Wco  = (const float*)d_in[21];
    const float* bj   = (const float*)d_in[22];
    const float* bi   = (const float*)d_in[23];
    const float* bf   = (const float*)d_in[24];
    const float* bo   = (const float*)d_in[25];
    const float* bc   = (const float*)d_in[26];
    const float* blast= (const float*)d_in[27];
    const float* btime= (const float*)d_in[28];
    const float* bd   = (const float*)d_in[29];
    float* out  = (float*)d_out;
    float* fwbt = (float*)d_ws;   // B_*T_ floats = 16 KB

    fw_precompute_kernel<<<(B_ * T_ + 255) / 256, 256, 0, stream>>>(freqs, fwbt);

    dim3 grid(F_, 2);   // 64 features x 2 directions = 128 independent recurrent chains
    timelstm_kernel<<<grid, 256, 0, stream>>>(
        inputs, times, lastv, fwbt,
        Uj, Ui, Uf, Uo, Uc, Ulast, Utime, Dw,
        Wj, Wi, Wf, Wo, Wc, Wd, Wdec,
        Wci, Wcf, Wco,
        bj, bi, bf, bo, bc, blast, btime, bd,
        out);
}
